// ANet_86036784873649
// MI455X (gfx1250) — matrix operand order repaired and checked by reading, not compiled, not run on hardware
//
#include <hip/hip_runtime.h>
#include <hip/hip_bf16.h>

// ---------------------------------------------------------------------------
// ANet on MI455X (gfx1250, wave32).  Bandwidth-bound: the two big GEMMs
// (128x128 conv over 256K points, and 256x128x131072 fc) run as
// v_wmma_f32_16x16x32_bf16.  LDS tiles are stored in *fragment order*
// ([tile][lane][16 bf16]) so each WMMA operand is two ds_load_b128.
// w_fc is streamed into LDS with double-buffered global_load_async_to_lds_b128
// (ASYNCcnt), overlapping DMA with WMMA.  BN2d+InstanceNorm collapse to a
// per-(b,c) affine; stage-1 activations are recomputed from x analytically.
// Workspace: ~157 MB (h2 tensor 134MB + split-K partials).
// ---------------------------------------------------------------------------

#define EPS_ 1e-5f
static constexpr int BB = 256;    // batch
static constexpr int PP_ = 1024;  // points
static constexpr int CC = 128;    // channels

typedef __attribute__((ext_vector_type(16))) __bf16 v16bf;
typedef __attribute__((ext_vector_type(8)))  float  v8f;

union Frag16 { v16bf v; uint4 q[2]; };

__device__ __forceinline__ unsigned short f2bf(float f) {
  unsigned u = __float_as_uint(f);
  u += 0x7FFFu + ((u >> 16) & 1u);   // round-to-nearest-even
  return (unsigned short)(u >> 16);
}
__device__ __forceinline__ unsigned packbf(float a, float b) {
  return (unsigned)f2bf(a) | ((unsigned)f2bf(b) << 16);
}

// Load one WMMA operand from fragment-ordered LDS: 32 contiguous bytes/lane.
__device__ __forceinline__ v16bf frag_ld(const unsigned short* p) {
  Frag16 f;
  const uint4* q = (const uint4*)p;
  f.q[0] = q[0];
  f.q[1] = q[1];
  return f.v;
}

// Inverse of the ISA 7.12.2 16-bit A-matrix (16x32) layout:
// element (row r, k) lives in lane r+16h at slot:  k<16: h=k>>3, slot=k&7;
// k>=16: h=(k-16)>>3, slot=8+((k-16)&7).   (consecutive even k -> same lane,
// adjacent slots, so pairs pack into one b32 store)
__device__ __forceinline__ int amap(int r, int kl, int& slot) {
  int h;
  if (kl < 16) { h = kl >> 3; slot = kl & 7; }
  else         { h = (kl - 16) >> 3; slot = 8 + ((kl - 16) & 7); }
  return r + 16 * h;
}
// B-matrix (32x16): element (k, n) -> lane n+16*((k>>3)&1), slot (k&7)+8*(k>>4).

__device__ __forceinline__ v8f wmma_bf16(v16bf a, v16bf b, v8f c) {
  return __builtin_amdgcn_wmma_f32_16x16x32_bf16(false, a, false, b,
                                                 (short)0, c, false, false);
}

// Async DMA: copy 16B from global to LDS (per active lane), ASYNCcnt-tracked.
__device__ __forceinline__ void async_ld_b128(unsigned lds_off,
                                              const void* gptr) {
  asm volatile("global_load_async_to_lds_b128 %0, %1, off"
               :: "v"(lds_off), "v"((unsigned long long)(uintptr_t)gptr)
               : "memory");
}

// ---------------------------------------------------------------------------
// K1: per-batch moments of x: Sx[3], Sxx[6] over 1024 points.
// ---------------------------------------------------------------------------
__global__ __launch_bounds__(256) void k1_xstats(const float* __restrict__ x,
                                                 int xoff,
                                                 float* __restrict__ s1) {
  __shared__ float red[256];
  const int b = blockIdx.x, t = threadIdx.x;
  float a[9];
#pragma unroll
  for (int q = 0; q < 9; ++q) a[q] = 0.f;
  for (int i = 0; i < 4; ++i) {
    int p = t + 256 * i;
    const float* xp = &x[(size_t)b * 6144 + xoff + (size_t)p * 3];
    float x0 = xp[0], x1 = xp[1], x2 = xp[2];
    a[0] += x0; a[1] += x1; a[2] += x2;
    a[3] += x0 * x0; a[4] += x0 * x1; a[5] += x0 * x2;
    a[6] += x1 * x1; a[7] += x1 * x2; a[8] += x2 * x2;
  }
  for (int q = 0; q < 9; ++q) {
    red[t] = a[q];
    __syncthreads();
    for (int s = 128; s > 0; s >>= 1) {
      if (t < s) red[t] += red[t + s];
      __syncthreads();
    }
    if (t == 0) s1[b * 9 + q] = red[0];
    __syncthreads();
  }
}

// ---------------------------------------------------------------------------
// K2: alpha1/beta1 per (b,c) for fused bn2d+inorm of h1 = W_in x + b_in.
// ---------------------------------------------------------------------------
__global__ __launch_bounds__(256) void k2_prep1(const float* __restrict__ s1,
                                                const float* __restrict__ w_in,
                                                const float* __restrict__ b_in,
                                                const float* __restrict__ g1,
                                                float* __restrict__ al1,
                                                float* __restrict__ be1) {
  __shared__ float tot[9];
  __shared__ float s1c[128];
  __shared__ float red[256];
  const int t = threadIdx.x;
  for (int q = 0; q < 9; ++q) {
    red[t] = s1[t * 9 + q];
    __syncthreads();
    for (int s = 128; s > 0; s >>= 1) {
      if (t < s) red[t] += red[t + s];
      __syncthreads();
    }
    if (t == 0) tot[q] = red[0];
    __syncthreads();
  }
  if (t < 128) {
    float w0 = w_in[t * 3], w1 = w_in[t * 3 + 1], w2 = w_in[t * 3 + 2];
    float bb = b_in[t];
    const float inv = 1.f / (256.f * 1024.f);
    float d = (w0 * tot[0] + w1 * tot[1] + w2 * tot[2]) * inv;
    float m = d + bb;
    float e2 = (w0 * w0 * tot[3] + 2.f * w0 * w1 * tot[4] +
                2.f * w0 * w2 * tot[5] + w1 * w1 * tot[6] +
                2.f * w1 * w2 * tot[7] + w2 * w2 * tot[8]) * inv +
               2.f * bb * d + bb * bb;
    float V = e2 - m * m;
    s1c[t] = g1[t] * rsqrtf(V + EPS_);   // bn scale; bn bias cancels in inorm
  }
  __syncthreads();
  for (int it = 0; it < 128; ++it) {
    int e = it * 256 + t;
    int b = e >> 7, c = e & 127;
    const float* sb = &s1[b * 9];
    float w0 = w_in[c * 3], w1 = w_in[c * 3 + 1], w2 = w_in[c * 3 + 2];
    float bb = b_in[c];
    const float invP = 1.f / 1024.f;
    float d = (w0 * sb[0] + w1 * sb[1] + w2 * sb[2]) * invP;
    float mu = d + bb;
    float e2 = (w0 * w0 * sb[3] + 2.f * w0 * w1 * sb[4] +
                2.f * w0 * w2 * sb[5] + w1 * w1 * sb[6] +
                2.f * w1 * w2 * sb[7] + w2 * w2 * sb[8]) * invP +
               2.f * bb * d + bb * bb;
    float var = e2 - mu * mu;
    float s = s1c[c];
    float al = s * rsqrtf(var * s * s + EPS_);
    al1[e] = al;
    be1[e] = -mu * al;
  }
}

// ---------------------------------------------------------------------------
// K3: h2 = W_c @ relu(al1*(W_in x + b_in) + be1) + b_c  via bf16 WMMA.
// W_c is swizzled once per block into fragment order (8 Mtiles x 4 Ktiles);
// the a1 tile (recomputed from x) is staged per K-step in fragment order.
// Also emits deterministic per-(b,c) partial sum/sumsq of h2 (shuffle-reduced).
// ---------------------------------------------------------------------------
__global__ __launch_bounds__(256) void k3_gemm1(
    const float* __restrict__ x, int xoff,
    const float* __restrict__ w_in, const float* __restrict__ b_in,
    const float* __restrict__ w_c, const float* __restrict__ b_c,
    const float* __restrict__ al1, const float* __restrict__ be1,
    float* __restrict__ h2, float* __restrict__ s2part) {
  __shared__ unsigned short sWcF[8 * 4 * 32 * 16];  // 32KB frag-ordered W_c
  __shared__ unsigned short sAF[4 * 32 * 16];       // 4KB frag-ordered a1 tile
  __shared__ float sX[64 * 3];
  __shared__ float sAl[128], sBe[128], sWin[128 * 3], sBin[128], sBc[128];

  const int pblk = blockIdx.x;   // 0..15
  const int b = blockIdx.y;      // 0..255
  const int t = threadIdx.x;
  const int p0 = pblk * 64;

  // W_c -> fragment order (pairs of consecutive c -> one b32 store)
  const float2* wc2 = (const float2*)w_c;
  for (int i = t; i < 8192; i += 256) {
    int o = i >> 6, c = (i & 63) * 2;
    float2 f = wc2[(o * 128 + c) >> 1];
    int slot;
    int ln = amap(o & 15, c & 31, slot);
    int mt = o >> 4, kt = c >> 5;
    ((unsigned*)sWcF)[(((mt * 4 + kt) * 32 + ln) * 16 + slot) >> 1] =
        packbf(f.x, f.y);
  }
  for (int i = t; i < 64 * 3; i += 256) {
    int p = i / 3, c3 = i % 3;
    sX[i] = x[(size_t)b * 6144 + xoff + (size_t)(p0 + p) * 3 + c3];
  }
  if (t < 128) {
    sAl[t] = al1[b * 128 + t];
    sBe[t] = be1[b * 128 + t];
    sBin[t] = b_in[t];
    sBc[t] = b_c[t];
  }
  for (int i = t; i < 128 * 3; i += 256) sWin[i] = w_in[i];
  __syncthreads();

  const int wid = t >> 5, lane = t & 31;
  v8f acc[4];
#pragma unroll
  for (int n = 0; n < 4; ++n)
#pragma unroll
    for (int e = 0; e < 8; ++e) acc[n][e] = 0.f;

  for (int kk = 0; kk < 4; ++kk) {
    // a1 tile (K=32 channels x N=64 points) -> fragment order
    for (int i = t; i < 1024; i += 256) {
      int kl = (i >> 6) * 2, p = i & 63;
      int c = kk * 32 + kl;
      float x0 = sX[p * 3], x1 = sX[p * 3 + 1], x2 = sX[p * 3 + 2];
      float ha = sWin[c * 3] * x0 + sWin[c * 3 + 1] * x1 +
                 sWin[c * 3 + 2] * x2 + sBin[c];
      float hb = sWin[c * 3 + 3] * x0 + sWin[c * 3 + 4] * x1 +
                 sWin[c * 3 + 5] * x2 + sBin[c + 1];
      float za = sAl[c] * ha + sBe[c];
      float zb = sAl[c + 1] * hb + sBe[c + 1];
      za = za > 0.f ? za : 0.f;
      zb = zb > 0.f ? zb : 0.f;
      int ln = (p & 15) + 16 * ((kl >> 3) & 1);   // B-layout: lane = n + 16*((k>>3)&1)
      int slot = (kl & 7) + 8 * (kl >> 4);
      ((unsigned*)sAF)[(((p >> 4) * 32 + ln) * 16 + slot) >> 1] =
          packbf(za, zb);
    }
    __syncthreads();
    v16bf afrag = frag_ld(&sWcF[((wid * 4 + kk) * 32 + lane) * 16]);
#pragma unroll
    for (int n = 0; n < 4; ++n) {
      v16bf bfrag = frag_ld(&sAF[(n * 32 + lane) * 16]);
      acc[n] = wmma_bf16(afrag, bfrag, acc[n]);
    }
    __syncthreads();
  }

  const int L = lane & 15, hh = lane >> 4;
#pragma unroll
  for (int r = 0; r < 8; ++r) {
    int row = wid * 16 + r + 8 * hh;
    float bias = sBc[row];
    float s = 0.f, ss = 0.f;
#pragma unroll
    for (int n = 0; n < 4; ++n) {
      float v = acc[n][r] + bias;
      h2[((size_t)b * 128 + row) * 1024 + p0 + n * 16 + L] = v;
      s += v;
      ss += v * v;
    }
#pragma unroll
    for (int m = 1; m < 16; m <<= 1) {   // reduce within 16-lane half
      s += __shfl_xor(s, m, 32);
      ss += __shfl_xor(ss, m, 32);
    }
    if (L == 0) {
      size_t o = (((size_t)b * 16 + pblk) * 128 + row) * 2;
      s2part[o] = s;
      s2part[o + 1] = ss;
    }
  }
}

// --------------------------- stage-2 norm prep -----------------------------
__global__ __launch_bounds__(256) void k4a_reduce(const float* __restrict__ s2p,
                                                  float* __restrict__ sbc,
                                                  float* __restrict__ ssbc) {
  int e = blockIdx.x * 256 + threadIdx.x;   // b*128+c
  int b = e >> 7, c = e & 127;
  float s = 0.f, ss = 0.f;
  for (int pb = 0; pb < 16; ++pb) {
    size_t o = (((size_t)b * 16 + pb) * 128 + c) * 2;
    s += s2p[o];
    ss += s2p[o + 1];
  }
  sbc[e] = s;
  ssbc[e] = ss;
}

__global__ __launch_bounds__(128) void k4b_chan(const float* __restrict__ sbc,
                                                const float* __restrict__ ssbc,
                                                const float* __restrict__ g2,
                                                float* __restrict__ sc2) {
  int c = threadIdx.x;
  float s = 0.f, ss = 0.f;
  for (int b = 0; b < 256; ++b) {
    s += sbc[b * 128 + c];
    ss += ssbc[b * 128 + c];
  }
  const float inv = 1.f / (256.f * 1024.f);
  float m = s * inv;
  float V = ss * inv - m * m;
  sc2[c] = g2[c] * rsqrtf(V + EPS_);
}

__global__ __launch_bounds__(256) void k4c_ab2(const float* __restrict__ sbc,
                                               const float* __restrict__ ssbc,
                                               const float* __restrict__ sc2,
                                               float* __restrict__ al2,
                                               float* __restrict__ be2) {
  int e = blockIdx.x * 256 + threadIdx.x;
  int c = e & 127;
  float mu = sbc[e] * (1.f / 1024.f);
  float var = ssbc[e] * (1.f / 1024.f) - mu * mu;
  float s = sc2[c];
  float al = s * rsqrtf(var * s * s + EPS_);
  al2[e] = al;
  be2[e] = -mu * al;
}

// ---------------------------------------------------------------------------
// K5: split-K fc GEMM.  f[b,o] = sum_{c,p} hf[b,c,p] * w_fc[o, c*P+p],
// hf = relu(al2*h2+be2) + relu(al1*(W_in x+b_in)+be1).
// Block = (c, mblk, oblk): 64 batches x 64 outputs, K-chunk = channel c.
// w_fc raw fp32 tiles stream through a double-buffered LDS ring via
// global_load_async_to_lds_b128 (ASYNCcnt), overlapping DMA with WMMA.
// ---------------------------------------------------------------------------
__global__ __launch_bounds__(256) void k5_fc(
    const float* __restrict__ x, int xoff,
    const float* __restrict__ h2, const float* __restrict__ w_fc,
    const float* __restrict__ w_in, const float* __restrict__ b_in,
    const float* __restrict__ al1, const float* __restrict__ be1,
    const float* __restrict__ al2, const float* __restrict__ be2,
    float* __restrict__ fpart) {
  __shared__ unsigned short sAtF[4 * 32 * 16];   // 4KB frag-ordered hf
  __shared__ unsigned short sBtF[4 * 32 * 16];   // 4KB frag-ordered w_fc
  __shared__ float sRaw[2][2048];                // 2x8KB raw w_fc ring
  __shared__ float sA1[64], sB1[64], sA2[64], sB2[64];

  const int c = blockIdx.x;          // channel chunk 0..127
  const int b0 = blockIdx.y * 64;
  const int o0 = blockIdx.z * 64;
  const int t = threadIdx.x;
  const float w0 = w_in[c * 3], w1 = w_in[c * 3 + 1], w2 = w_in[c * 3 + 2];
  const float bi = b_in[c];

  if (t < 64) {
    int b = b0 + t;
    sA1[t] = al1[b * 128 + c];
    sB1[t] = be1[b * 128 + c];
    sA2[t] = al2[b * 128 + c];
    sB2[t] = be2[b * 128 + c];
  }

  // async copy of one 64o x 32p fp32 w_fc tile into sRaw[buf]
  auto issueB = [&](int step, int buf) {
#pragma unroll
    for (int k = 0; k < 2; ++k) {
      int j = t + 256 * k;            // 512 x 16B chunks
      int oi = j >> 3, pi = (j & 7) * 4;
      const float* g = &w_fc[(size_t)(o0 + oi) * 131072 + (size_t)c * 1024 +
                             step * 32 + pi];
      unsigned lds = (unsigned)(uintptr_t)&sRaw[buf][0] + (unsigned)(j * 16);
      async_ld_b128(lds, g);
    }
  };
  issueB(0, 0);

  const int wid = t >> 5, lane = t & 31;
  const int m = wid & 3, op = wid >> 2;
  v8f acc0, acc1;
#pragma unroll
  for (int e = 0; e < 8; ++e) { acc0[e] = 0.f; acc1[e] = 0.f; }

  for (int step = 0; step < 32; ++step) {
    const int p0 = step * 32;
    if (step + 1 < 32) {
      issueB(step + 1, (step + 1) & 1);
      asm volatile("s_wait_asynccnt 0x2" ::: "memory");  // current tile done
    } else {
      asm volatile("s_wait_asynccnt 0x0" ::: "memory");
    }
    __syncthreads();   // everyone's async data visible

    // convert raw w_fc -> fragment-ordered bf16 (B: k=p, n=o)
    for (int i = t; i < 1024; i += 256) {
      int oi = i >> 4, kl = (i & 15) * 2;
      float2 wv = ((const float2*)&sRaw[step & 1][0])[(oi * 32 + kl) >> 1];
      int ln = (oi & 15) + 16 * ((kl >> 3) & 1);
      int slot = (kl & 7) + 8 * (kl >> 4);
      ((unsigned*)sBtF)[(((oi >> 4) * 32 + ln) * 16 + slot) >> 1] =
          packbf(wv.x, wv.y);
    }
    // stage hf (A: m=batch, k=p) -> fragment order
    for (int i = t; i < 1024; i += 256) {
      int bl = i >> 4, kl = (i & 15) * 2;
      int b = b0 + bl, p = p0 + kl;
      const float* hp = &h2[((size_t)b * 128 + c) * 1024 + p];
      float2 hv = *(const float2*)hp;
      __builtin_prefetch(hp + 32, 0, 1);      // next step -> global_prefetch_b8
      float hf0 = hv.x * sA2[bl] + sB2[bl];
      float hf1 = hv.y * sA2[bl] + sB2[bl];
      hf0 = hf0 > 0.f ? hf0 : 0.f;
      hf1 = hf1 > 0.f ? hf1 : 0.f;
      const float* xp = &x[(size_t)b * 6144 + xoff + (size_t)p * 3];
      float ha = w0 * xp[0] + w1 * xp[1] + w2 * xp[2] + bi;
      float hb = w0 * xp[3] + w1 * xp[4] + w2 * xp[5] + bi;
      float a1v = ha * sA1[bl] + sB1[bl];
      float b1v = hb * sA1[bl] + sB1[bl];
      a1v = a1v > 0.f ? a1v : 0.f;
      b1v = b1v > 0.f ? b1v : 0.f;
      int slot;
      int ln = amap(bl & 15, kl, slot);
      ((unsigned*)sAtF)[(((bl >> 4) * 32 + ln) * 16 + slot) >> 1] =
          packbf(hf0 + a1v, hf1 + b1v);
    }
    __syncthreads();
    v16bf a = frag_ld(&sAtF[(m * 32 + lane) * 16]);
    v16bf bb0 = frag_ld(&sBtF[((2 * op) * 32 + lane) * 16]);
    v16bf bb1 = frag_ld(&sBtF[((2 * op + 1) * 32 + lane) * 16]);
    acc0 = wmma_bf16(a, bb0, acc0);
    acc1 = wmma_bf16(a, bb1, acc1);
    __syncthreads();
  }

  const int L = lane & 15, hh = lane >> 4;
#pragma unroll
  for (int r = 0; r < 8; ++r) {
    int bg = b0 + m * 16 + r + 8 * hh;
    fpart[((size_t)c * 256 + bg) * 128 + o0 + (2 * op) * 16 + L] = acc0[r];
    fpart[((size_t)c * 256 + bg) * 128 + o0 + (2 * op + 1) * 16 + L] = acc1[r];
  }
}

__global__ __launch_bounds__(256) void k6_reduce(const float* __restrict__ fp,
                                                 const float* __restrict__ bfc,
                                                 float* __restrict__ Y,
                                                 int colOff) {
  int e = blockIdx.x * 256 + threadIdx.x;   // 32768 = b*128+o
  int b = e >> 7, o = e & 127;
  float s = bfc[o];
  for (int cc = 0; cc < 128; ++cc) s += fp[((size_t)cc * 256 + b) * 128 + o];
  Y[b * 256 + colOff + o] = s;
}

// ------------------------------- head --------------------------------------
__global__ __launch_bounds__(256) void k7_l1(const float* __restrict__ Y,
                                             const float* __restrict__ w1,
                                             const float* __restrict__ bh1,
                                             float* __restrict__ Z1) {
  __shared__ float row[256];
  int b = blockIdx.x, j = threadIdx.x;
  row[j] = Y[b * 256 + j];
  __syncthreads();
  float s = bh1[j];
  for (int k = 0; k < 256; ++k) s += row[k] * w1[j * 256 + k];
  Z1[b * 256 + j] = s;
}

__global__ void k8_bnstats(const float* __restrict__ Z,
                           const float* __restrict__ g,
                           const float* __restrict__ be,
                           float* __restrict__ ab, int F) {
  int j = threadIdx.x;
  if (j >= F) return;
  float s = 0.f, ss = 0.f;
  for (int b = 0; b < 256; ++b) {
    float v = Z[b * F + j];
    s += v;
    ss += v * v;
  }
  float m = s * (1.f / 256.f);
  float V = ss * (1.f / 256.f) - m * m;
  float a = g[j] * rsqrtf(V + EPS_);
  ab[2 * j] = a;
  ab[2 * j + 1] = be[j] - m * a;
}

__global__ __launch_bounds__(128) void k9_l2(const float* __restrict__ Z1,
                                             const float* __restrict__ ab,
                                             const float* __restrict__ w2,
                                             const float* __restrict__ bh2,
                                             float* __restrict__ Z2) {
  __shared__ float act[256];
  int b = blockIdx.x, j = threadIdx.x;
  for (int k = j; k < 256; k += 128) {
    float v = Z1[b * 256 + k] * ab[2 * k] + ab[2 * k + 1];
    act[k] = v > 0.f ? v : 0.f;
  }
  __syncthreads();
  float s = bh2[j];
  for (int k = 0; k < 256; ++k) s += act[k] * w2[j * 256 + k];
  Z2[b * 128 + j] = s;
}

// final 16-dim linear + symmetric 4x4 Jacobi eigensolver, one thread per row.
__global__ __launch_bounds__(256) void k11_head(const float* __restrict__ Z2,
                                                const float* __restrict__ ab,
                                                const float* __restrict__ w3,
                                                const float* __restrict__ bh3,
                                                float* __restrict__ out) {
  int b = threadIdx.x;
  float y[16];
  for (int o = 0; o < 16; ++o) {
    float s = bh3[o];
    for (int k = 0; k < 128; ++k) {
      float v = Z2[b * 128 + k] * ab[2 * k] + ab[2 * k + 1];
      v = v > 0.f ? v : 0.f;
      s += v * w3[o * 128 + k];
    }
    y[o] = s;
  }
  float A[4][4], V[4][4];
  for (int i = 0; i < 4; ++i)
    for (int j = 0; j < 4; ++j) {
      A[i][j] = 0.5f * (y[i * 4 + j] + y[j * 4 + i]);
      V[i][j] = (i == j) ? 1.f : 0.f;
    }
  const int Pi[6] = {0, 0, 0, 1, 1, 2};
  const int Qi[6] = {1, 2, 3, 2, 3, 3};
  for (int sweep = 0; sweep < 15; ++sweep) {
    for (int r = 0; r < 6; ++r) {
      int p = Pi[r], q = Qi[r];
      float apq = A[p][q];
      if (fabsf(apq) > 1e-20f) {
        float th = (A[q][q] - A[p][p]) / (2.f * apq);
        float tt = (th >= 0.f ? 1.f : -1.f) /
                   (fabsf(th) + sqrtf(th * th + 1.f));
        float cc = rsqrtf(tt * tt + 1.f);
        float s_ = tt * cc;
        for (int k = 0; k < 4; ++k) {   // columns: A <- A*J
          float akp = A[k][p], akq = A[k][q];
          A[k][p] = cc * akp - s_ * akq;
          A[k][q] = s_ * akp + cc * akq;
        }
        for (int k = 0; k < 4; ++k) {   // rows: A <- J^T*A
          float apk = A[p][k], aqk = A[q][k];
          A[p][k] = cc * apk - s_ * aqk;
          A[q][k] = s_ * apk + cc * aqk;
        }
        for (int k = 0; k < 4; ++k) {   // V <- V*J
          float vkp = V[k][p], vkq = V[k][q];
          V[k][p] = cc * vkp - s_ * vkq;
          V[k][q] = s_ * vkp + cc * vkq;
        }
      }
    }
  }
  int imin = 0;
  float emin = A[0][0];
  for (int i = 1; i < 4; ++i)
    if (A[i][i] < emin) { emin = A[i][i]; imin = i; }
  float q0 = V[0][imin];
  float sgn = (q0 < 0.f) ? -1.f : 1.f;
  for (int k = 0; k < 4; ++k) out[b * 4 + k] = sgn * V[k][imin];
}

// ---------------------------------------------------------------------------
extern "C" void kernel_launch(void* const* d_in, const int* in_sizes, int n_in,
                              void* d_out, int out_size, void* d_ws,
                              size_t ws_size, hipStream_t stream) {
  (void)in_sizes; (void)n_in; (void)out_size; (void)ws_size;
  const float* x = (const float*)d_in[0];
  const float* w1h = (const float*)d_in[21];
  const float* bh1 = (const float*)d_in[22];
  const float* gb1 = (const float*)d_in[23];
  const float* beb1 = (const float*)d_in[24];
  const float* w2h = (const float*)d_in[25];
  const float* bh2 = (const float*)d_in[26];
  const float* gb2 = (const float*)d_in[27];
  const float* beb2 = (const float*)d_in[28];
  const float* w3h = (const float*)d_in[29];
  const float* bh3 = (const float*)d_in[30];
  float* out = (float*)d_out;

  float* w = (float*)d_ws;
  size_t o = 0;
  float* wsH2 = w + o;   o += (size_t)BB * CC * PP_;          // 33.5M
  float* wsFP = w + o;   o += (size_t)128 * 256 * 128;        // 4.2M
  float* wsS2 = w + o;   o += (size_t)256 * 16 * 128 * 2;     // 1.0M
  float* wsSBC = w + o;  o += 32768;
  float* wsSSB = w + o;  o += 32768;
  float* wsS1 = w + o;   o += 2304;
  float* wsSC2 = w + o;  o += 128;
  float* wsAL1 = w + o;  o += 32768;
  float* wsBE1 = w + o;  o += 32768;
  float* wsAL2 = w + o;  o += 32768;
  float* wsBE2 = w + o;  o += 32768;
  float* wsY = w + o;    o += 65536;
  float* wsZ1 = w + o;   o += 65536;
  float* wsZ2 = w + o;   o += 32768;
  float* wsAB1 = w + o;  o += 512;
  float* wsAB2 = w + o;  o += 256;
  // total ~39.2M floats (~157 MB) of workspace

  for (int net = 0; net < 2; ++net) {
    const int base = 1 + net * 10;
    const float* w_in = (const float*)d_in[base + 0];
    const float* b_in = (const float*)d_in[base + 1];
    const float* g1 = (const float*)d_in[base + 2];
    const float* w_c = (const float*)d_in[base + 4];
    const float* b_c = (const float*)d_in[base + 5];
    const float* g2 = (const float*)d_in[base + 6];
    const float* w_fc = (const float*)d_in[base + 8];
    const float* b_fc = (const float*)d_in[base + 9];
    const int xoff = net * 3072;

    k1_xstats<<<BB, 256, 0, stream>>>(x, xoff, wsS1);
    k2_prep1<<<1, 256, 0, stream>>>(wsS1, w_in, b_in, g1, wsAL1, wsBE1);
    k3_gemm1<<<dim3(16, BB), 256, 0, stream>>>(x, xoff, w_in, b_in, w_c, b_c,
                                               wsAL1, wsBE1, wsH2, wsS2);
    k4a_reduce<<<128, 256, 0, stream>>>(wsS2, wsSBC, wsSSB);
    k4b_chan<<<1, 128, 0, stream>>>(wsSBC, wsSSB, g2, wsSC2);
    k4c_ab2<<<128, 256, 0, stream>>>(wsSBC, wsSSB, wsSC2, wsAL2, wsBE2);
    k5_fc<<<dim3(128, 4, 2), 256, 0, stream>>>(x, xoff, wsH2, w_fc, w_in,
                                               b_in, wsAL1, wsBE1, wsAL2,
                                               wsBE2, wsFP);
    k6_reduce<<<128, 256, 0, stream>>>(wsFP, b_fc, wsY, net * 128);
  }

  k7_l1<<<256, 256, 0, stream>>>(wsY, w1h, bh1, wsZ1);
  k8_bnstats<<<1, 256, 0, stream>>>(wsZ1, gb1, beb1, wsAB1, 256);
  k9_l2<<<256, 128, 0, stream>>>(wsZ1, wsAB1, w2h, bh2, wsZ2);
  k8_bnstats<<<1, 128, 0, stream>>>(wsZ2, gb2, beb2, wsAB2, 128);
  k11_head<<<1, 256, 0, stream>>>(wsZ2, wsAB2, w3h, bh3, out);
}
